// HighFreqMonteCarloLagrangianMLS_69312182222949
// MI455X (gfx1250) — matrix operand order repaired and checked by reading, not compiled, not run on hardware
//
#include <hip/hip_runtime.h>
#include <hip/hip_bf16.h>
#include <stdint.h>

#define RG 128
#define KPATHS 8
#define SIGMA 0.8f
#define GRID_CH 8          // 7 used channels + 1 pad -> 32B cells

typedef __attribute__((ext_vector_type(16))) _Float16 v16h;
typedef __attribute__((ext_vector_type(8)))  _Float16 v8h;
typedef __attribute__((ext_vector_type(8)))  float    v8f;

// Branch-free tanh-GELU: 0.5*x*(1+tanh(y)) == x * sigmoid(2y) == x/(1+e^(-2y))
__device__ __forceinline__ float gelu_tanh(float x) {
    const float c0 = 0.7978845608028654f;  // sqrt(2/pi)
    float y = c0 * (x + 0.044715f * x * x * x);
    return x * __frcp_rn(1.0f + __expf(-2.0f * y));
}

__device__ __forceinline__ uint32_t pcg(uint32_t v) {
    v = v * 747796405u + 2891336453u;
    uint32_t w = ((v >> ((v >> 28u) + 4u)) ^ v) * 277803737u;
    return (w >> 22u) ^ w;
}

// ---------------------------------------------------------------- zero grid
__global__ void zero_grid_kernel(float4* g, int n4) {
    int i = blockIdx.x * blockDim.x + threadIdx.x;
    int stride = gridDim.x * blockDim.x;
    float4 z; z.x = z.y = z.z = z.w = 0.0f;
    for (; i < n4; i += stride) g[i] = z;
}

// ------------------------------------------------- pack weights to B-layout
// B fragment packing used by this code (16-bit, 32x16): lane<16 holds K=0..15, lane>=16 holds
// K=16..31; column N = lane&15. Stored contiguously: frag*512 + lane*16 + i.
__global__ void prep_weights_kernel(const float* __restrict__ W1,
                                    const float* __restrict__ W2,
                                    const float* __restrict__ W3,
                                    _Float16* pw1, _Float16* pw2, _Float16* pw3) {
    int t = threadIdx.x;
    // W_d1: (24x64) -> padded K=32, 4 N-tiles
    for (int idx = t; idx < 4 * 512; idx += blockDim.x) {
        int frag = idx >> 9, lane = (idx >> 4) & 31, i = idx & 15;
        int nt = frag, hs = lane >> 4;
        int col = nt * 16 + (lane & 15);
        int k = hs * 16 + i;
        float v = (k < 24) ? W1[k * 64 + col] : 0.0f;
        pw1[idx] = (_Float16)v;
    }
    // W_d2: (64x64) -> 2 K-tiles x 4 N-tiles
    for (int idx = t; idx < 8 * 512; idx += blockDim.x) {
        int frag = idx >> 9, lane = (idx >> 4) & 31, i = idx & 15;
        int kt = frag >> 2, nt = frag & 3, hs = lane >> 4;
        int col = nt * 16 + (lane & 15);
        int k = kt * 32 + hs * 16 + i;
        pw2[idx] = (_Float16)W2[k * 64 + col];
    }
    // W_d3: (64x3) -> padded N=16, 2 K-tiles
    for (int idx = t; idx < 2 * 512; idx += blockDim.x) {
        int frag = idx >> 9, lane = (idx >> 4) & 31, i = idx & 15;
        int hs = lane >> 4;
        int col = lane & 15;
        int k = frag * 32 + hs * 16 + i;
        float v = (col < 3) ? W3[k * 3 + col] : 0.0f;
        pw3[idx] = (_Float16)v;
    }
}

// ----------------------------------------------------------------- P2G scatter
__global__ void p2g_kernel(const float* __restrict__ xsr,
                           const float* __restrict__ xsc,
                           const float* __restrict__ Wf,
                           const float* __restrict__ bf,
                           float* __restrict__ grid, int N) {
    int i = blockIdx.x * blockDim.x + threadIdx.x;
    if (i >= N) return;
    float xs[3], u[3];
#pragma unroll
    for (int c = 0; c < 3; ++c) {
        xs[c] = fminf(fmaxf(xsr[3 * i + c], 0.0f), 1.0f);
        u[c]  = xsc[3 * i + c] - xs[c];
    }
    float content[7];
#pragma unroll
    for (int j = 0; j < 3; ++j) {
        float a = u[0] * Wf[0 * 3 + j] + u[1] * Wf[1 * 3 + j] + u[2] * Wf[2 * 3 + j] + bf[j];
        content[j] = gelu_tanh(a);
    }
    content[3] = 1.0f;
    content[4] = u[0]; content[5] = u[1]; content[6] = u[2];

    float cx[3]; int bx[3]; float dx[3];
#pragma unroll
    for (int c = 0; c < 3; ++c) {
        cx[c] = fminf(fmaxf(xs[c] * (float)(RG - 1), 0.0f), (float)(RG - 1) - 0.001f);
        bx[c] = (int)floorf(cx[c]);
        dx[c] = cx[c] - (float)bx[c];
    }
#pragma unroll
    for (int c = 0; c < 8; ++c) {
        int ox = (c >> 2) & 1, oy = (c >> 1) & 1, oz = c & 1;
        float w = (ox ? dx[0] : 1.0f - dx[0]) *
                  (oy ? dx[1] : 1.0f - dx[1]) *
                  (oz ? dx[2] : 1.0f - dx[2]);
        long flat = ((long)(bx[0] + ox) * RG + (bx[1] + oy)) * RG + (bx[2] + oz);
        float* cell = grid + flat * GRID_CH;
#pragma unroll
        for (int ch = 0; ch < 7; ++ch) atomicAdd(cell + ch, content[ch] * w);
    }
}

// ------------------------------------------- MC gather + normalize + features
__global__ void gather_kernel(const float* __restrict__ xqr,
                              const float* __restrict__ grid,
                              _Float16* __restrict__ dec,
                              float* __restrict__ unorm, int M) {
    int m = blockIdx.x * blockDim.x + threadIdx.x;
    if (m >= M) return;
    float xq[3], q[3];
#pragma unroll
    for (int c = 0; c < 3; ++c) {
        xq[c] = fminf(fmaxf(xqr[3 * m + c], 0.0f), 1.0f);
        q[c]  = xq[c] * (float)(RG - 1);
    }
    float acc[7];
#pragma unroll
    for (int j = 0; j < 7; ++j) acc[j] = 0.0f;

    for (int k = 0; k < KPATHS; ++k) {
        uint32_t s = (uint32_t)m * 8u + (uint32_t)k;
        // Box-Muller: 3 normals from 4 uniforms (counter-hash PRNG)
        float u1 = ((float)pcg(s * 4u + 0u) + 1.0f) * 2.3283064365386963e-10f;
        float u2 = (float)pcg(s * 4u + 1u) * 2.3283064365386963e-10f;
        float u3 = ((float)pcg(s * 4u + 2u) + 1.0f) * 2.3283064365386963e-10f;
        float u4 = (float)pcg(s * 4u + 3u) * 2.3283064365386963e-10f;
        float r1 = __fsqrt_rn(-2.0f * __logf(u1)), a1 = 6.2831853071795864f * u2;
        float r2 = __fsqrt_rn(-2.0f * __logf(u3)), a2 = 6.2831853071795864f * u4;
        float n[3];
        n[0] = r1 * __cosf(a1);
        n[1] = r1 * __sinf(a1);
        n[2] = r2 * __cosf(a2);

        float p[3]; int lo[3]; float fr[3]; int i0[3], i1[3];
#pragma unroll
        for (int c = 0; c < 3; ++c) {
            p[c]  = q[c] + SIGMA * n[c];
            lo[c] = (int)floorf(p[c]);
            fr[c] = p[c] - (float)lo[c];
            i0[c] = min(max(lo[c], 0), RG - 1);
            i1[c] = min(max(lo[c] + 1, 0), RG - 1);
        }
#pragma unroll
        for (int c = 0; c < 8; ++c) {
            int ox = (c >> 2) & 1, oy = (c >> 1) & 1, oz = c & 1;
            int ix = ox ? i1[0] : i0[0];
            int iy = oy ? i1[1] : i0[1];
            int iz = oz ? i1[2] : i0[2];
            float w = (ox ? fr[0] : 1.0f - fr[0]) *
                      (oy ? fr[1] : 1.0f - fr[1]) *
                      (oz ? fr[2] : 1.0f - fr[2]);
            const float* cell = grid + (((long)ix * RG + iy) * RG + iz) * GRID_CH;
            float4 va = *(const float4*)cell;
            float4 vb = *(const float4*)(cell + 4);
            acc[0] += w * va.x; acc[1] += w * va.y; acc[2] += w * va.z;
            acc[3] += w * va.w; acc[4] += w * vb.x; acc[5] += w * vb.y;
            acc[6] += w * vb.z;
        }
    }
#pragma unroll
    for (int j = 0; j < 7; ++j) acc[j] *= (1.0f / (float)KPATHS);

    float denom = fmaxf(acc[3], 1e-5f);
    float mask  = (acc[3] > 1e-5f) ? 1.0f : 0.0f;
    float inv   = mask * __frcp_rn(denom);

    _Float16 buf[32];
#pragma unroll
    for (int c = 0; c < 3; ++c) {
        buf[c]     = (_Float16)(acc[c] * inv);       // f_norm
        float un   = acc[4 + c] * inv;               // u_norm
        buf[3 + c] = (_Float16)un;
        unorm[(size_t)m * 4 + c] = un;
    }
#pragma unroll
    for (int L = 0; L < 3; ++L) {
        float freq = (float)(1 << L) * 3.14159265358979f;
#pragma unroll
        for (int c = 0; c < 3; ++c) {
            buf[6 + L * 6 + c]     = (_Float16)__sinf(freq * xq[c]);
            buf[6 + L * 6 + 3 + c] = (_Float16)__cosf(freq * xq[c]);
        }
    }
#pragma unroll
    for (int j = 24; j < 32; ++j) buf[j] = (_Float16)0.0f;

    v8h* drow = (v8h*)(dec + (size_t)m * 32);
#pragma unroll
    for (int v = 0; v < 4; ++v) {
        v8h tmp;
#pragma unroll
        for (int e = 0; e < 8; ++e) tmp[e] = buf[v * 8 + e];
        drow[v] = tmp;
    }
}

// ----------------------------------------------------- WMMA decoder MLP
// One wave processes a 16-query tile; 14 v_wmma_f32_16x16x32_f16 per tile.
__launch_bounds__(256)
__global__ void mlp_wmma_kernel(const _Float16* __restrict__ dec,
                                const float* __restrict__ unorm,
                                const _Float16* __restrict__ pw1,
                                const _Float16* __restrict__ pw2,
                                const _Float16* __restrict__ pw3,
                                const float* __restrict__ b1,
                                const float* __restrict__ b2,
                                const float* __restrict__ b3,
                                float* __restrict__ out, int M) {
    __shared__ _Float16 lds1[8][16][64];
    __shared__ _Float16 lds2[8][16][64];

    int wv = threadIdx.x >> 5;
    int l  = threadIdx.x & 31;
    int hs = l >> 4;         // half-wave select
    int ln = l & 15;

    int tile = blockIdx.x * 8 + wv;
    int base = tile * 16;
    if (base >= M) return;   // uniform per wave: EXEC stays all-ones

    // ---- A1 fragment (16x32 f16): lane<16 -> K 0..15; lane>=16 -> K 16..31
    int row = base + ln; if (row >= M) row = M - 1;
    const _Float16* dr = dec + (size_t)row * 32;
    v8h alo = *(const v8h*)(dr + hs * 16);
    v8h ahi = *(const v8h*)(dr + hs * 16 + 8);
    v16h a1;
#pragma unroll
    for (int i = 0; i < 8; ++i) { a1[i] = alo[i]; a1[8 + i] = ahi[i]; }

    // ---- layer 1: 24(->32) -> 64, 4 WMMAs
#pragma unroll
    for (int nt = 0; nt < 4; ++nt) {
        v16h b = *(const v16h*)(pw1 + (((nt * 32) + l) << 4));
        float bias = b1[nt * 16 + ln];
        v8f c;
#pragma unroll
        for (int r = 0; r < 8; ++r) c[r] = bias;
        c = __builtin_amdgcn_wmma_f32_16x16x32_f16(false, a1, false, b, (short)0, c, false, false);
#pragma unroll
        for (int r = 0; r < 8; ++r)
            lds1[wv][hs * 8 + r][nt * 16 + ln] = (_Float16)gelu_tanh(c[r]);
    }
    __builtin_amdgcn_wave_barrier();

    // ---- A2 fragments from LDS (D-layout -> A-layout transpose)
    v16h a2[2];
#pragma unroll
    for (int kt = 0; kt < 2; ++kt) {
        const _Float16* hr = &lds1[wv][ln][kt * 32];
        v8h p0 = *(const v8h*)(hr + hs * 16);
        v8h p1 = *(const v8h*)(hr + hs * 16 + 8);
#pragma unroll
        for (int i = 0; i < 8; ++i) { a2[kt][i] = p0[i]; a2[kt][8 + i] = p1[i]; }
    }

    // ---- layer 2: 64 -> 64, 8 WMMAs
#pragma unroll
    for (int nt = 0; nt < 4; ++nt) {
        float bias = b2[nt * 16 + ln];
        v8f c;
#pragma unroll
        for (int r = 0; r < 8; ++r) c[r] = bias;
        v16h bk0 = *(const v16h*)(pw2 + (((0 * 4 + nt) * 32 + l) << 4));
        c = __builtin_amdgcn_wmma_f32_16x16x32_f16(false, a2[0], false, bk0, (short)0, c, false, false);
        v16h bk1 = *(const v16h*)(pw2 + (((1 * 4 + nt) * 32 + l) << 4));
        c = __builtin_amdgcn_wmma_f32_16x16x32_f16(false, a2[1], false, bk1, (short)0, c, false, false);
#pragma unroll
        for (int r = 0; r < 8; ++r)
            lds2[wv][hs * 8 + r][nt * 16 + ln] = (_Float16)gelu_tanh(c[r]);
    }
    __builtin_amdgcn_wave_barrier();

    // ---- A3 fragments
    v16h a3[2];
#pragma unroll
    for (int kt = 0; kt < 2; ++kt) {
        const _Float16* hr = &lds2[wv][ln][kt * 32];
        v8h p0 = *(const v8h*)(hr + hs * 16);
        v8h p1 = *(const v8h*)(hr + hs * 16 + 8);
#pragma unroll
        for (int i = 0; i < 8; ++i) { a3[kt][i] = p0[i]; a3[kt][8 + i] = p1[i]; }
    }

    // ---- layer 3: 64 -> 3(->16), 2 WMMAs
    float bias3 = (ln < 3) ? b3[ln] : 0.0f;
    v8f c3;
#pragma unroll
    for (int r = 0; r < 8; ++r) c3[r] = bias3;
    {
        v16h b0 = *(const v16h*)(pw3 + ((0 * 32 + l) << 4));
        c3 = __builtin_amdgcn_wmma_f32_16x16x32_f16(false, a3[0], false, b0, (short)0, c3, false, false);
        v16h b1f = *(const v16h*)(pw3 + ((1 * 32 + l) << 4));
        c3 = __builtin_amdgcn_wmma_f32_16x16x32_f16(false, a3[1], false, b1f, (short)0, c3, false, false);
    }

    // ---- epilogue: out = clip(u_norm + res, 0.001, 0.999)
    if (ln < 3) {
#pragma unroll
        for (int r = 0; r < 8; ++r) {
            int rr = base + hs * 8 + r;
            if (rr < M) {
                float u = unorm[(size_t)rr * 4 + ln];
                float v = u + c3[r];
                out[(size_t)rr * 3 + ln] = fminf(fmaxf(v, 0.001f), 0.999f);
            }
        }
    }
}

// -------------------------------------------------------------- launcher
extern "C" void kernel_launch(void* const* d_in, const int* in_sizes, int n_in,
                              void* d_out, int out_size, void* d_ws, size_t ws_size,
                              hipStream_t stream) {
    const float* xq  = (const float*)d_in[0];
    const float* xsr = (const float*)d_in[1];
    const float* xsc = (const float*)d_in[2];
    const float* Wf  = (const float*)d_in[3];
    const float* bf  = (const float*)d_in[4];
    const float* W1  = (const float*)d_in[5];
    const float* b1  = (const float*)d_in[6];
    const float* W2  = (const float*)d_in[7];
    const float* b2  = (const float*)d_in[8];
    const float* W3  = (const float*)d_in[9];
    const float* b3  = (const float*)d_in[10];
    int M = in_sizes[0] / 3;
    int N = in_sizes[1] / 3;

    char* ws = (char*)d_ws;
    size_t off = 0;
    float* grid = (float*)(ws + off);
    off += (size_t)RG * RG * RG * GRID_CH * sizeof(float);          // 64 MB
    _Float16* dec = (_Float16*)(ws + off);
    off += (size_t)M * 32 * sizeof(_Float16);
    off = (off + 255) & ~(size_t)255;
    float* un = (float*)(ws + off);
    off += (size_t)M * 4 * sizeof(float);
    off = (off + 255) & ~(size_t)255;
    _Float16* pw1 = (_Float16*)(ws + off); off += 4 * 512 * sizeof(_Float16);
    _Float16* pw2 = (_Float16*)(ws + off); off += 8 * 512 * sizeof(_Float16);
    _Float16* pw3 = (_Float16*)(ws + off); off += 2 * 512 * sizeof(_Float16);
    (void)ws_size; (void)n_in; (void)out_size;

    int n4 = (RG * RG * RG * GRID_CH) / 4;
    zero_grid_kernel<<<2048, 256, 0, stream>>>((float4*)grid, n4);
    prep_weights_kernel<<<1, 256, 0, stream>>>(W1, W2, W3, pw1, pw2, pw3);
    p2g_kernel<<<(N + 255) / 256, 256, 0, stream>>>(xsr, xsc, Wf, bf, grid, N);
    gather_kernel<<<(M + 255) / 256, 256, 0, stream>>>(xq, grid, dec, un, M);
    int tiles = (M + 15) / 16;
    mlp_wmma_kernel<<<(tiles + 7) / 8, 256, 0, stream>>>(
        dec, un, pw1, pw2, pw3, b1, b2, b3, (float*)d_out, M);
}
